// TransformerEncoderBlock_27393301413917
// MI455X (gfx1250) — compile-verified, matrix operand order checked by reading, not run on hardware
//
#include <hip/hip_runtime.h>
#include <math.h>

// Model dims (fixed by reference)
#define L_   6
#define B_   2
#define S_   1024
#define D_   512
#define H_   8
#define DK_  64
#define DFF_ 2048
#define M_   (B_ * S_)   // 2048 rows of activations

typedef unsigned short bf16_t;                                    // raw bf16 bits in memory
typedef __attribute__((ext_vector_type(16))) __bf16 v16bf;        // WMMA A/B operand
typedef __attribute__((ext_vector_type(8)))  float  v8f;          // WMMA C/D accumulator

#define USE_ASYNC_LDS 1

// ---------------------------------------------------------------- helpers
__device__ __forceinline__ bf16_t f2bf(float f) {
    union { float f; unsigned u; } v; v.f = f;
    unsigned r = v.u + 0x7FFFu + ((v.u >> 16) & 1u);  // round-to-nearest-even
    return (bf16_t)(r >> 16);
}

// Load the per-lane 16 bf16 values of a WMMA 16-bit A/B operand.
// p points at (row_base + klo): elements 0..7 = K klo..klo+7, 8..15 = K klo+16..klo+23.
__device__ __forceinline__ v16bf load_ab(const bf16_t* p) {
    union { v16bf v; uint4 u[2]; } t;
    t.u[0] = *reinterpret_cast<const uint4*>(p);
    t.u[1] = *reinterpret_cast<const uint4*>(p + 16);
    return t.v;
}

__device__ __forceinline__ v8f wmma_bf16(v16bf a, v16bf b, v8f c) {
    return __builtin_amdgcn_wmma_f32_16x16x32_bf16(false, a, false, b, (short)0, c, false, false);
}

#if USE_ASYNC_LDS
// Async 16B copy global -> LDS via the VGLOBAL async encoding (ASYNCcnt-tracked).
// vdst = 32-bit LDS byte address (wave-relative), vaddr = 64-bit global address.
__device__ __forceinline__ void async_copy_b128(const bf16_t* gsrc, bf16_t* ldst) {
    const unsigned lds_addr =
        (unsigned)(size_t)(__attribute__((address_space(3))) void*)(void*)ldst;
    asm volatile("global_load_async_to_lds_b128 %0, %1, off"
                 :
                 : "v"(lds_addr), "v"((unsigned long long)(size_t)gsrc)
                 : "memory");
}
__device__ __forceinline__ void async_wait0() {
    __builtin_amdgcn_s_wait_asynccnt(0);
}
#endif

// ---------------------------------------------------------------- generic WMMA GEMM
// C[M,N] = A[M,K] (bf16 row-major) x B (bf16, provided as Bt[N,K] row-major).
// Block = 128 threads = 4 waves stacked in M; each wave computes a 64x32 C tile
// (4x2 WMMA tiles).  The 32x32 B tile of the K-step is shared by all waves and is
// staged in LDS via GLOBAL_LOAD_ASYNC_TO_LDS_B128 (ASYNCcnt).
// OUT_MODE: 0 f32; 1 bf16; 2 relu->bf16; 3 bf16 scattered to vT[(b*H+h)*DK+d][s]
// (requires N == D_); 4 f32 AND bf16.
#define BROW_PAD 48   // LDS row stride in bf16 elements (96B: 16B-aligned, bank-spread)

template <int OUT_MODE>
__global__ __launch_bounds__(128) void gemm_wmma(const bf16_t* __restrict__ A,
                                                 const bf16_t* __restrict__ Bt,
                                                 float* __restrict__ Cf,
                                                 bf16_t* __restrict__ Cb,
                                                 int M, int N, int K) {
    __shared__ bf16_t Bs[32 * BROW_PAD];

    const int tid  = threadIdx.x;
    const int wave = tid >> 5;
    const int lane = tid & 31;
    const int r    = lane & 15;
    const int klo  = (lane >> 4) << 3;              // 0 or 8 (ISA 16-bit A/B layout)

    const int tn  = blockIdx.x * 32;
    const int tm0 = blockIdx.y * 256 + wave * 64;

    const bf16_t* Ap = A + (size_t)(tm0 + r) * K + klo;   // + mi*16*K per m-subtile

    // B staging assignment: thread copies one 16B chunk per K-step.
    const int brow = tid >> 2;                       // 0..31 (n within tile)
    const int bseg = tid & 3;                        // 0..3  (16B chunk within 64B row)
    const bf16_t* Bg = Bt + (size_t)(tn + brow) * K + bseg * 8;
    bf16_t*       Bl = Bs + brow * BROW_PAD + bseg * 8;

    v8f acc[8] = {};                                 // [mi][nj], idx = mi*2 + nj

    for (int k0 = 0; k0 < K; k0 += 32) {
        // ---- stage 32x32 B tile into LDS
#if USE_ASYNC_LDS
        async_copy_b128(Bg + k0, Bl);
        async_wait0();
#else
        *reinterpret_cast<uint4*>(Bl) = *reinterpret_cast<const uint4*>(Bg + k0);
#endif
        __syncthreads();

        // ---- read B operands from LDS (lane layout)
        v16bf b0 = load_ab(Bs + (size_t)(0 * 16 + r) * BROW_PAD + klo);
        v16bf b1 = load_ab(Bs + (size_t)(1 * 16 + r) * BROW_PAD + klo);

        // ---- A operands straight from global (per-wave private)
        __builtin_prefetch(Ap + k0 + 512, 0, 1);     // global_prefetch_b8
        v16bf a0 = load_ab(Ap + (size_t)0 * 16 * K + k0);
        v16bf a1 = load_ab(Ap + (size_t)1 * 16 * K + k0);
        v16bf a2 = load_ab(Ap + (size_t)2 * 16 * K + k0);
        v16bf a3 = load_ab(Ap + (size_t)3 * 16 * K + k0);

        acc[0] = wmma_bf16(a0, b0, acc[0]);
        acc[1] = wmma_bf16(a0, b1, acc[1]);
        acc[2] = wmma_bf16(a1, b0, acc[2]);
        acc[3] = wmma_bf16(a1, b1, acc[3]);
        acc[4] = wmma_bf16(a2, b0, acc[4]);
        acc[5] = wmma_bf16(a2, b1, acc[5]);
        acc[6] = wmma_bf16(a3, b0, acc[6]);
        acc[7] = wmma_bf16(a3, b1, acc[7]);

        __syncthreads();                              // B tile consumed before overwrite
    }

#pragma unroll
    for (int mi = 0; mi < 4; ++mi) {
#pragma unroll
        for (int nj = 0; nj < 2; ++nj) {
            const v8f av    = acc[mi * 2 + nj];
            const int n     = tn + nj * 16 + (lane & 15);
            const int mbase = tm0 + mi * 16 + ((lane >> 4) << 3);
#pragma unroll
            for (int i = 0; i < 8; ++i) {
                const int m = mbase + i;
                const float v = av[i];
                if (OUT_MODE == 0) {
                    Cf[(size_t)m * N + n] = v;
                } else if (OUT_MODE == 1) {
                    Cb[(size_t)m * N + n] = f2bf(v);
                } else if (OUT_MODE == 2) {
                    Cb[(size_t)m * N + n] = f2bf(v > 0.f ? v : 0.f);
                } else if (OUT_MODE == 3) {
                    const int b = m / S_, s = m % S_;
                    const int h = n / DK_, d = n % DK_;
                    Cb[((size_t)((b * H_ + h) * DK_ + d)) * S_ + s] = f2bf(v);
                } else {  // 4
                    Cf[(size_t)m * N + n] = v;
                    Cb[(size_t)m * N + n] = f2bf(v);
                }
            }
        }
    }
}

// ---------------------------------------------------------------- attention scores
// scores[b,h,q,k] = (Q Kt)/8 with mask==0 -> -1e9 ; written straight into d_out attn
// slice.  One wave per 32x32 tile (2x2 WMMA register blocking).
__global__ __launch_bounds__(32) void attn_scores(const bf16_t* __restrict__ qb,
                                                  const bf16_t* __restrict__ kb,
                                                  const int* __restrict__ mask,
                                                  float* __restrict__ attn_out) {
    const int bh   = blockIdx.z;
    const int b    = bh / H_, h = bh % H_;
    const int tn   = blockIdx.x * 32;   // key positions
    const int tm   = blockIdx.y * 32;   // query positions
    const int lane = threadIdx.x & 31;
    const int r    = lane & 15;
    const int klo  = (lane >> 4) << 3;

    const bf16_t* Ap = qb + (size_t)(b * S_ + tm + r) * D_ + h * DK_ + klo;
    const bf16_t* Bp = kb + (size_t)(b * S_ + tn + r) * D_ + h * DK_ + klo;
    const size_t  mstride = (size_t)16 * D_;

    v8f acc[4] = {};
#pragma unroll
    for (int k0 = 0; k0 < DK_; k0 += 32) {
        v16bf a0 = load_ab(Ap + k0);
        v16bf a1 = load_ab(Ap + mstride + k0);
        v16bf b0 = load_ab(Bp + k0);
        v16bf b1 = load_ab(Bp + mstride + k0);
        acc[0] = wmma_bf16(a0, b0, acc[0]);
        acc[1] = wmma_bf16(a0, b1, acc[1]);
        acc[2] = wmma_bf16(a1, b0, acc[2]);
        acc[3] = wmma_bf16(a1, b1, acc[3]);
    }

    float*     outp = attn_out + (size_t)bh * S_ * S_;
    const int* mrow = mask + (size_t)b * S_ * S_;
#pragma unroll
    for (int mi = 0; mi < 2; ++mi) {
#pragma unroll
        for (int nj = 0; nj < 2; ++nj) {
            const v8f av    = acc[mi * 2 + nj];
            const int n     = tn + nj * 16 + (lane & 15);
            const int mbase = tm + mi * 16 + ((lane >> 4) << 3);
#pragma unroll
            for (int i = 0; i < 8; ++i) {
                const int m = mbase + i;
                float v = av[i] * 0.125f;  // 1/sqrt(64)
                if (mrow[(size_t)m * S_ + n] == 0) v = -1e9f;
                outp[(size_t)m * S_ + n] = v;
            }
        }
    }
}

// ---------------------------------------------------------------- softmax (row len S_)
__global__ __launch_bounds__(256) void softmax_rows(float* __restrict__ attn,
                                                    bf16_t* __restrict__ attnb) {
    const size_t base = (size_t)blockIdx.x * S_;
    float*  p  = attn  + base;
    bf16_t* pb = attnb + base;
    const int t = threadIdx.x;
    __shared__ float red[256];

    float l0 = p[t], l1 = p[t + 256], l2 = p[t + 512], l3 = p[t + 768];
    red[t] = fmaxf(fmaxf(l0, l1), fmaxf(l2, l3));
    __syncthreads();
    for (int s = 128; s > 0; s >>= 1) { if (t < s) red[t] = fmaxf(red[t], red[t + s]); __syncthreads(); }
    const float rmax = red[0];
    __syncthreads();

    float e0 = expf(l0 - rmax), e1 = expf(l1 - rmax), e2 = expf(l2 - rmax), e3 = expf(l3 - rmax);
    red[t] = e0 + e1 + e2 + e3;
    __syncthreads();
    for (int s = 128; s > 0; s >>= 1) { if (t < s) red[t] += red[t + s]; __syncthreads(); }
    const float inv = 1.0f / red[0];

    const float o0 = e0 * inv, o1 = e1 * inv, o2 = e2 * inv, o3 = e3 * inv;
    p[t] = o0; p[t + 256] = o1; p[t + 512] = o2; p[t + 768] = o3;
    pb[t] = f2bf(o0); pb[t + 256] = f2bf(o1); pb[t + 512] = f2bf(o2); pb[t + 768] = f2bf(o3);
}

// ---------------------------------------------------------------- attn @ V
// o[b,s,h*DK+d] = sum_k attn[b,h,s,k] * v[b,k,h*DK+d]  (V pre-stored transposed:
// vT[(bh*DK+d)][k]).  One wave per 32x32 tile (2x2 WMMA register blocking).
__global__ __launch_bounds__(32) void attn_av(const bf16_t* __restrict__ attnb,
                                              const bf16_t* __restrict__ vT,
                                              float* __restrict__ o) {
    const int bh   = blockIdx.z;
    const int b    = bh / H_, h = bh % H_;
    const int tn   = blockIdx.x * 32;   // over DK
    const int tm   = blockIdx.y * 32;   // over S
    const int lane = threadIdx.x & 31;
    const int r    = lane & 15;
    const int klo  = (lane >> 4) << 3;

    const bf16_t* Ap = attnb + (size_t)bh * S_ * S_  + (size_t)(tm + r) * S_ + klo;
    const bf16_t* Bp = vT    + (size_t)bh * DK_ * S_ + (size_t)(tn + r) * S_ + klo;
    const size_t  rstride = (size_t)16 * S_;

    v8f acc[4] = {};
    for (int k0 = 0; k0 < S_; k0 += 32) {
        __builtin_prefetch(Ap + k0 + 512, 0, 1);
        v16bf a0 = load_ab(Ap + k0);
        v16bf a1 = load_ab(Ap + rstride + k0);
        v16bf b0 = load_ab(Bp + k0);
        v16bf b1 = load_ab(Bp + rstride + k0);
        acc[0] = wmma_bf16(a0, b0, acc[0]);
        acc[1] = wmma_bf16(a0, b1, acc[1]);
        acc[2] = wmma_bf16(a1, b0, acc[2]);
        acc[3] = wmma_bf16(a1, b1, acc[3]);
    }

#pragma unroll
    for (int mi = 0; mi < 2; ++mi) {
#pragma unroll
        for (int nj = 0; nj < 2; ++nj) {
            const v8f av    = acc[mi * 2 + nj];
            const int n     = tn + nj * 16 + (lane & 15);
            const int mbase = tm + mi * 16 + ((lane >> 4) << 3);
#pragma unroll
            for (int i = 0; i < 8; ++i) {
                const int m = mbase + i;
                o[(size_t)(b * S_ + m) * D_ + h * DK_ + n] = av[i];
            }
        }
    }
}

// ---------------------------------------------------------------- LayerNorm over D_=512
template <bool HAS_RES, bool OUT_F32, bool OUT_BF16>
__global__ __launch_bounds__(256) void layernorm_k(const float* __restrict__ x,
                                                   const float* __restrict__ res,
                                                   const float* __restrict__ g,
                                                   const float* __restrict__ be,
                                                   float* __restrict__ of,
                                                   bf16_t* __restrict__ ob) {
    const size_t base = (size_t)blockIdx.x * D_;
    const int t = threadIdx.x;
    __shared__ float red[256];

    float v0 = x[base + t], v1 = x[base + t + 256];
    if (HAS_RES) { v0 += res[base + t]; v1 += res[base + t + 256]; }

    red[t] = v0 + v1;
    __syncthreads();
    for (int s = 128; s > 0; s >>= 1) { if (t < s) red[t] += red[t + s]; __syncthreads(); }
    const float mu = red[0] * (1.0f / D_);
    __syncthreads();

    const float d0 = v0 - mu, d1 = v1 - mu;
    red[t] = d0 * d0 + d1 * d1;
    __syncthreads();
    for (int s = 128; s > 0; s >>= 1) { if (t < s) red[t] += red[t + s]; __syncthreads(); }
    const float rstd = rsqrtf(red[0] * (1.0f / D_) + 1e-5f);

    const float o0 = d0 * rstd * g[t]       + be[t];
    const float o1 = d1 * rstd * g[t + 256] + be[t + 256];
    if (OUT_F32)  { of[base + t] = o0; of[base + t + 256] = o1; }
    if (OUT_BF16) { ob[base + t] = f2bf(o0); ob[base + t + 256] = f2bf(o1); }
}

// ---------------------------------------------------------------- positional encoding add
__global__ __launch_bounds__(256) void add_pe(const float* __restrict__ xin,
                                              float* __restrict__ xout) {
    const size_t idx = (size_t)blockIdx.x * 256 + threadIdx.x;
    if (idx >= (size_t)M_ * D_) return;
    const int d  = (int)(idx % D_);
    const int s  = (int)((idx / D_) % S_);
    const int i2 = (d >> 1) << 1;
    const float ang = (float)s * expf((float)i2 * (-9.210340371976184f / (float)D_));
    const float pe  = (d & 1) ? cosf(ang) : sinf(ang);
    xout[idx] = xin[idx] + pe;
}

// ---------------------------------------------------------------- W[K,N] f32 -> Wt[N,K] bf16
__global__ __launch_bounds__(256) void conv_t_bf16(const float* __restrict__ W,
                                                   bf16_t* __restrict__ Wt,
                                                   int K, int N) {
    const size_t idx = (size_t)blockIdx.x * 256 + threadIdx.x;
    if (idx >= (size_t)K * N) return;
    const int n = (int)(idx % N);
    const int k = (int)(idx / N);
    Wt[(size_t)n * K + k] = f2bf(W[idx]);
}

// ================================================================ launch
extern "C" void kernel_launch(void* const* d_in, const int* in_sizes, int n_in,
                              void* d_out, int out_size, void* d_ws, size_t ws_size,
                              hipStream_t stream) {
    const float* x_in     = (const float*)d_in[0];
    const int*   mask     = (const int*)d_in[1];
    const float* Wq       = (const float*)d_in[2];
    const float* Wk       = (const float*)d_in[3];
    const float* Wv       = (const float*)d_in[4];
    const float* Wfc      = (const float*)d_in[5];
    const float* ln_mha_g = (const float*)d_in[6];
    const float* ln_mha_b = (const float*)d_in[7];
    const float* W1       = (const float*)d_in[8];
    const float* W2       = (const float*)d_in[9];
    const float* ln_ff_g  = (const float*)d_in[10];
    const float* ln_ff_b  = (const float*)d_in[11];

    float* out_x    = (float*)d_out;
    float* out_attn = out_x + (size_t)M_ * D_;   // [L,B,H,S,S]

    // ---- workspace carve (256B aligned)
    char* wp = (char*)d_ws;
    auto carve = [&](size_t bytes) -> char* {
        char* p = wp; wp += (bytes + 255) & ~(size_t)255; return p;
    };
    float*  xbuf  = (float*) carve((size_t)M_ * D_ * 4);
    bf16_t* hb    = (bf16_t*)carve((size_t)M_ * D_ * 2);
    bf16_t* qb    = (bf16_t*)carve((size_t)M_ * D_ * 2);
    bf16_t* kb    = (bf16_t*)carve((size_t)M_ * D_ * 2);
    bf16_t* vT    = (bf16_t*)carve((size_t)M_ * D_ * 2);
    bf16_t* atnb  = (bf16_t*)carve((size_t)B_ * H_ * S_ * S_ * 2);
    float*  obuf  = (float*) carve((size_t)M_ * D_ * 4);
    bf16_t* olnb  = (bf16_t*)carve((size_t)M_ * D_ * 2);
    float*  o2f   = (float*) carve((size_t)M_ * D_ * 4);
    bf16_t* o2b   = (bf16_t*)carve((size_t)M_ * D_ * 2);
    bf16_t* ffhb  = (bf16_t*)carve((size_t)M_ * DFF_ * 2);
    float*  fff   = (float*) carve((size_t)M_ * D_ * 4);
    bf16_t* Wqt   = (bf16_t*)carve((size_t)D_ * D_ * 2);
    bf16_t* Wkt   = (bf16_t*)carve((size_t)D_ * D_ * 2);
    bf16_t* Wvt   = (bf16_t*)carve((size_t)D_ * D_ * 2);
    bf16_t* Wfct  = (bf16_t*)carve((size_t)D_ * D_ * 2);
    bf16_t* W1t   = (bf16_t*)carve((size_t)D_ * DFF_ * 2);  // [DFF, D]
    bf16_t* W2t   = (bf16_t*)carve((size_t)DFF_ * D_ * 2);  // [D, DFF]

    const dim3 blk128(128), blk256(256), blk32(32);

    // x = x + PE
    add_pe<<<(M_ * D_ + 255) / 256, blk256, 0, stream>>>(x_in, xbuf);

    for (int l = 0; l < L_; ++l) {
        // weight conversion (transpose to [N,K] bf16)
        const size_t wDD = (size_t)D_ * D_, wDF = (size_t)D_ * DFF_;
        conv_t_bf16<<<(int)((wDD + 255) / 256), blk256, 0, stream>>>(Wq  + (size_t)l * wDD, Wqt,  D_,  D_);
        conv_t_bf16<<<(int)((wDD + 255) / 256), blk256, 0, stream>>>(Wk  + (size_t)l * wDD, Wkt,  D_,  D_);
        conv_t_bf16<<<(int)((wDD + 255) / 256), blk256, 0, stream>>>(Wv  + (size_t)l * wDD, Wvt,  D_,  D_);
        conv_t_bf16<<<(int)((wDD + 255) / 256), blk256, 0, stream>>>(Wfc + (size_t)l * wDD, Wfct, D_,  D_);
        conv_t_bf16<<<(int)((wDF + 255) / 256), blk256, 0, stream>>>(W1  + (size_t)l * wDF, W1t,  D_,  DFF_);
        conv_t_bf16<<<(int)((wDF + 255) / 256), blk256, 0, stream>>>(W2  + (size_t)l * wDF, W2t,  DFF_, D_);

        const float* g1 = ln_mha_g + (size_t)l * D_;
        const float* b1 = ln_mha_b + (size_t)l * D_;
        const float* g2 = ln_ff_g  + (size_t)l * D_;
        const float* b2 = ln_ff_b  + (size_t)l * D_;

        // h = LN(x) -> bf16
        layernorm_k<false, false, true><<<M_, blk256, 0, stream>>>(xbuf, nullptr, g1, b1, nullptr, hb);

        // Q, K, V projections (V stored transposed per head)
        gemm_wmma<1><<<dim3(D_ / 32, M_ / 256), blk128, 0, stream>>>(hb, Wqt, nullptr, qb, M_, D_, D_);
        gemm_wmma<1><<<dim3(D_ / 32, M_ / 256), blk128, 0, stream>>>(hb, Wkt, nullptr, kb, M_, D_, D_);
        gemm_wmma<3><<<dim3(D_ / 32, M_ / 256), blk128, 0, stream>>>(hb, Wvt, nullptr, vT, M_, D_, D_);

        // attention
        float* attn_l = out_attn + (size_t)l * B_ * H_ * S_ * S_;
        attn_scores<<<dim3(S_ / 32, S_ / 32, B_ * H_), blk32, 0, stream>>>(qb, kb, mask, attn_l);
        softmax_rows<<<B_ * H_ * S_, blk256, 0, stream>>>(attn_l, atnb);
        attn_av<<<dim3(DK_ / 32, S_ / 32, B_ * H_), blk32, 0, stream>>>(atnb, vT, obuf);

        // o2 = LN(o) @ Wfc   (need f32 for residual and bf16 for FFN input)
        layernorm_k<false, false, true><<<M_, blk256, 0, stream>>>(obuf, nullptr, g1, b1, nullptr, olnb);
        gemm_wmma<4><<<dim3(D_ / 32, M_ / 256), blk128, 0, stream>>>(olnb, Wfct, o2f, o2b, M_, D_, D_);

        // FFN: ffh = relu(o2 @ W1) ; ff = ffh @ W2
        gemm_wmma<2><<<dim3(DFF_ / 32, M_ / 256), blk128, 0, stream>>>(o2b, W1t, nullptr, ffhb, M_, DFF_, D_);
        gemm_wmma<0><<<dim3(D_ / 32, M_ / 256), blk128, 0, stream>>>(ffhb, W2t, fff, nullptr, M_, D_, DFF_);

        // x = LN(o2 + ff)
        float* xdst = (l == L_ - 1) ? out_x : xbuf;
        layernorm_k<true, true, false><<<M_, blk256, 0, stream>>>(o2f, fff, g2, b2, xdst, nullptr);
    }
}
